// GatedDeltaNet_25512105738537
// MI455X (gfx1250) — matrix operand order repaired and checked by reading, not compiled, not run on hardware
//
#include <hip/hip_runtime.h>

// ---------------- problem constants ----------------
#define BB    2
#define TT    2048
#define DIMM  2048
#define HVV   32
#define HKK   16
#define DKK   128
#define DVV   128
#define KWW   4
#define KEYD  2048           // HK*DK
#define VALD  4096           // HV*DV
#define CONVD 8192           // 2*KEYD + VALD
#define EPSF  1e-6f
#define BTOT  (BB*TT)        // 4096 rows

typedef __attribute__((ext_vector_type(16))) __bf16 v16bf;
typedef __attribute__((ext_vector_type(8)))  __bf16 v8bf;
typedef __attribute__((ext_vector_type(8)))  float  v8f;

#if defined(__has_builtin)
#if __has_builtin(__builtin_amdgcn_tensor_load_to_lds) && __has_builtin(__builtin_amdgcn_s_wait_tensorcnt)
#define GDN_TDM 1
#endif
#endif
#ifndef GDN_TDM
#define GDN_TDM 0
#endif

// ---------------- cast helpers ----------------
__global__ void gdn_cast_bf16(const float* __restrict__ in, __bf16* __restrict__ out, size_t n) {
    size_t i = (size_t)blockIdx.x * blockDim.x + threadIdx.x;
    if (i < n) out[i] = (__bf16)in[i];
}

// in[R,C] fp32 (row-major) -> out[C,R] bf16 (row-major), i.e. transpose
__global__ void gdn_castT_bf16(const float* __restrict__ in, __bf16* __restrict__ out,
                               int R, int C) {
    size_t i = (size_t)blockIdx.x * blockDim.x + threadIdx.x;
    size_t n = (size_t)R * C;
    if (i >= n) return;
    int r = (int)(i / C);
    int c = (int)(i % C);
    out[(size_t)c * R + r] = (__bf16)in[i];
}

__device__ inline v16bf gdn_frag16(const __bf16* p_lo, const __bf16* p_hi) {
    uint4 p0 = *(const uint4*)p_lo;
    uint4 p1 = *(const uint4*)p_hi;
    v8bf a0 = __builtin_bit_cast(v8bf, p0);
    v8bf a1 = __builtin_bit_cast(v8bf, p1);
    return __builtin_shufflevector(a0, a1, 0,1,2,3,4,5,6,7,8,9,10,11,12,13,14,15);
}

// ---------------- WMMA GEMM: C[M,N] = A[M,K](bf16) * Bt[N,K](bf16)^T ----------------
// 256 threads = 8 waves (2M x 4N); wave tile 64x32 (4x2 WMMA tiles) -> block tile 128x128.
// Requires M%128==0, N%128==0, K%32==0 (true for all three GEMMs) -> no divergence.
template <typename CT>
__global__ __launch_bounds__(256) void gdn_gemm_bf16(const __bf16* __restrict__ A,
                                                     const __bf16* __restrict__ Bt,
                                                     CT* __restrict__ C,
                                                     int M, int N, int K) {
    const int lane  = threadIdx.x & 31;
    const int wid   = threadIdx.x >> 5;      // 0..7
    const int half  = lane >> 4;             // 0/1
    const int l16   = lane & 15;
    const int tileM = blockIdx.y * 128 + (wid >> 2) * 64;   // wave covers rows [tileM, tileM+64)
    const int tileN = blockIdx.x * 128 + (wid & 3) * 32;    // wave covers cols [tileN, tileN+32)

    const __bf16* __restrict__ ar[4];
#pragma unroll
    for (int im = 0; im < 4; ++im)
        ar[im] = A + (size_t)(tileM + im * 16 + l16) * K;
    const __bf16* __restrict__ br[2];
#pragma unroll
    for (int in = 0; in < 2; ++in)
        br[in] = Bt + (size_t)(tileN + in * 16 + l16) * K;

    v8f acc[4][2] = {};
    for (int k0 = 0; k0 < K; k0 += 32) {
        v16bf av[4], bv[2];
#pragma unroll
        for (int im = 0; im < 4; ++im)
            // A 16x32 layout: lanes 0-15 hold K={0..7,16..23}, lanes 16-31 hold {8..15,24..31}
            av[im] = gdn_frag16(ar[im] + k0 + (half << 3), ar[im] + k0 + 16 + (half << 3));
#pragma unroll
        for (int in = 0; in < 2; ++in)
            // B 32x16 layout: lanes 0-15 hold K={0..7,16..23}, lanes 16-31 hold {8..15,24..31}
            bv[in] = gdn_frag16(br[in] + k0 + (half << 3), br[in] + k0 + 16 + (half << 3));
#pragma unroll
        for (int im = 0; im < 4; ++im)
#pragma unroll
            for (int in = 0; in < 2; ++in)
                acc[im][in] = __builtin_amdgcn_wmma_f32_16x16x32_bf16(
                    false, av[im], false, bv[in], (short)0, acc[im][in], false, false);
    }
#pragma unroll
    for (int im = 0; im < 4; ++im)
#pragma unroll
        for (int in = 0; in < 2; ++in)
#pragma unroll
            for (int r = 0; r < 8; ++r) {
                int row = tileM + im * 16 + r + (half << 3);
                int col = tileN + in * 16 + l16;
                C[(size_t)row * N + col] = (CT)acc[im][in][r];
            }
}

// ---------------- depthwise causal conv (KW=4) + SiLU ----------------
__global__ void gdn_conv_silu(const __bf16* __restrict__ mixed,   // [B*T, CONVD] bf16
                              const float* __restrict__ conv_w,   // [CONVD, KW]
                              float* __restrict__ out) {          // [B*T, CONVD] f32
    size_t idx = (size_t)blockIdx.x * blockDim.x + threadIdx.x;
    if (idx >= (size_t)BTOT * CONVD) return;
    int c = (int)(idx % CONVD);
    size_t bt = idx / CONVD;
    int t = (int)(bt % TT);
    size_t base = (bt - (size_t)t) * CONVD + c;   // row (b, 0)
    float acc = 0.f;
#pragma unroll
    for (int i = 0; i < KWW; ++i) {
        int ts = t + i - (KWW - 1);
        if (ts >= 0) acc += (float)mixed[base + (size_t)ts * CONVD] * conv_w[c * KWW + i];
    }
    out[idx] = acc / (1.f + __expf(-acc));        // silu
}

// ---------------- l2norm(q)*DK^-0.5, l2norm(k) in-place ----------------
__global__ void gdn_prep_qk(float* __restrict__ qc) {             // [B*T, CONVD]
    int idx = blockIdx.x * blockDim.x + threadIdx.x;              // over BTOT*HKK
    if (idx >= BTOT * HKK) return;
    int hk = idx % HKK;
    size_t bt = (size_t)(idx / HKK);
    float* q = qc + bt * CONVD + (size_t)hk * DKK;
    float* k = q + KEYD;
    float sq = 0.f, sk = 0.f;
    for (int j = 0; j < DKK; ++j) { sq += q[j] * q[j]; sk += k[j] * k[j]; }
    float rq = rsqrtf(sq + EPSF) * 0.08838834764831845f;          // * DK^-0.5
    float rk = rsqrtf(sk + EPSF);
    for (int j = 0; j < DKK; ++j) { q[j] *= rq; k[j] *= rk; }
}

// ---------------- a/b projections -> g, beta ----------------
__global__ void gdn_gate(const float* __restrict__ x, const float* __restrict__ Wa,
                         const float* __restrict__ Wb, const float* __restrict__ dt_bias,
                         const float* __restrict__ A_log,
                         float* __restrict__ g, float* __restrict__ beta) {
    int idx = blockIdx.x * blockDim.x + threadIdx.x;              // over BTOT*HVV
    if (idx >= BTOT * HVV) return;
    int h = idx % HVV;
    const float* xr = x + (size_t)(idx / HVV) * DIMM;
    float sa = 0.f, sb = 0.f;
    for (int i = 0; i < DIMM; ++i) {
        float xv = xr[i];
        sa += xv * Wa[(size_t)i * HVV + h];
        sb += xv * Wb[(size_t)i * HVV + h];
    }
    beta[idx] = 1.f / (1.f + __expf(-sb));
    float sp_in = sa + dt_bias[h];
    float sp = (sp_in > 20.f) ? sp_in : log1pf(__expf(sp_in));    // softplus
    g[idx] = -__expf(A_log[h]) * sp;
}

#if GDN_TDM
typedef __attribute__((ext_vector_type(4))) unsigned int gdn_u32x4;
typedef __attribute__((ext_vector_type(8))) int          gdn_i32x8;
typedef __attribute__((ext_vector_type(4))) int          gdn_i32x4;
typedef __attribute__((address_space(3)))   float        gdn_lds_f;

// TDM: load 2 rows of 128 f32 (q then k), row stride 2048 elements, into LDS (contiguous).
__device__ inline void gdn_tdm_load_qk(const float* gptr, unsigned lds_byte) {
    unsigned long long ga = (unsigned long long)gptr;
    gdn_u32x4 g0 = { 1u,                                   // count=1, flags=0
                     lds_byte,                             // lds_addr
                     (unsigned)(ga & 0xFFFFFFFFull),       // global_addr[31:0]
                     (unsigned)((ga >> 32) & 0x1FFFFFFull) // global_addr[56:32]
                     | 0x80000000u };                      // type=2 ("image")
    gdn_i32x8 g1 = { (int)0x00020000u,        // wg_mask=0, data_size=2 (4B)
                     (int)(2048u << 16),      // tensor_dim0 lo16 = 2048
                     (int)(4u << 16),         // tensor_dim0 hi=0 | tensor_dim1 lo16 = 4
                     (int)(128u << 16),       // tensor_dim1 hi=0 | tile_dim0 = 128
                     (int)2u,                 // tile_dim1 = 2, tile_dim2 = 0
                     (int)2048,               // tensor_dim0_stride lo32 = 2048
                     0,                       // stride0 hi16 = 0 | tensor_dim1_stride lo16 = 0
                     0 };
    gdn_i32x4 gz4 = { 0, 0, 0, 0 };
    gdn_i32x8 gz8 = { 0, 0, 0, 0, 0, 0, 0, 0 };
    // 6-arg toolchain variant: (g0, g1, g2, g3, extra, cpol)
    __builtin_amdgcn_tensor_load_to_lds(g0, g1, gz4, gz4, gz8, 0);
}
#endif

// ---------------- gated delta-rule scan ----------------
// grid = B*HV blocks, 128 threads; thread j owns state column S[:,j] in VGPRs.
// q/k slices for step t+1 are prefetched into an LDS double buffer by the TDM
// (wave 0 issues; completion gated with s_wait_tensorcnt + barrier); v,g,beta
// are register-prefetched.  Hides per-step global latency behind the FMA body.
__global__ __launch_bounds__(128) void gdn_scan(const float* __restrict__ qc,
                                                const float* __restrict__ g,
                                                const float* __restrict__ beta,
                                                float* __restrict__ o) {
    const int bh = blockIdx.x;           // b*HV + h
    const int b  = bh / HVV;
    const int h  = bh % HVV;
    const int hk = h >> 1;               // repeat factor HV/HK = 2
    const int j  = threadIdx.x;          // 0..127

    __shared__ float qk[2][2 * DKK];     // [buf][0..127]=q, [128..255]=k

    float S[DKK];
#pragma unroll
    for (int i = 0; i < DKK; ++i) S[i] = 0.f;

    const size_t bt0 = (size_t)b * TT;
    const float* row0 = qc + bt0 * CONVD;

#if GDN_TDM
    unsigned qk_lds0 = (unsigned)(unsigned long long)(gdn_lds_f*)&qk[0][0];
    if (threadIdx.x < 32)
        gdn_tdm_load_qk(row0 + (size_t)hk * DKK, qk_lds0);
#else
    float rqv = row0[(size_t)hk * DKK + j];
    float rkv = row0[KEYD + (size_t)hk * DKK + j];
#endif
    float rv = row0[2 * KEYD + (size_t)h * DVV + j];
    float rg = g[bt0 * HVV + h];
    float rb = beta[bt0 * HVV + h];

    for (int t = 0; t < TT; ++t) {
        const int buf = t & 1;
#if GDN_TDM
        if (threadIdx.x < 32) __builtin_amdgcn_s_wait_tensorcnt(0);
        __syncthreads();
#else
        __syncthreads();
        qk[buf][j]       = rqv;
        qk[buf][DKK + j] = rkv;
        __syncthreads();
#endif
        float vt  = rv;
        float e   = __expf(rg);
        float bt_ = rb;

        if (t + 1 < TT) {                                 // prefetch next step
            size_t btn = bt0 + t + 1;
            const float* nrow = qc + btn * CONVD;
#if GDN_TDM
            if (threadIdx.x < 32)
                gdn_tdm_load_qk(nrow + (size_t)hk * DKK,
                                qk_lds0 + (unsigned)((buf ^ 1) * 2 * DKK * 4));
#else
            rqv = nrow[(size_t)hk * DKK + j];
            rkv = nrow[KEYD + (size_t)hk * DKK + j];
#endif
            rv = nrow[2 * KEYD + (size_t)h * DVV + j];
            rg = g[btn * HVV + h];
            rb = beta[btn * HVV + h];
        }

        const float* lq = &qk[buf][0];
        const float* lk = &qk[buf][DKK];
        float pred = 0.f;
#pragma unroll
        for (int i = 0; i < DKK; ++i) {
            S[i] *= e;
            pred = fmaf(lk[i], S[i], pred);
        }
        float u  = (vt - pred) * bt_;
        float oj = 0.f;
#pragma unroll
        for (int i = 0; i < DKK; ++i) {
            S[i] = fmaf(lk[i], u, S[i]);
            oj = fmaf(lq[i], S[i], oj);
        }
        o[((bt0 + t) * HVV + h) * (size_t)DVV + j] = oj;
    }
}

// ---------------- gated RMSNorm * silu(z) -> bf16 ----------------
__global__ void gdn_norm_gate(const float* __restrict__ o, const float* __restrict__ z,
                              const float* __restrict__ nw, __bf16* __restrict__ onb) {
    int idx = blockIdx.x * blockDim.x + threadIdx.x;              // over BTOT*HVV
    if (idx >= BTOT * HVV) return;
    const float* orow = o + (size_t)idx * DVV;
    const float* zrow = z + (size_t)idx * DVV;    // z contiguous [B*T, HV*DV]
    float var = 0.f;
    for (int jj = 0; jj < DVV; ++jj) var += orow[jj] * orow[jj];
    float r = rsqrtf(var * (1.f / DVV) + EPSF);
    for (int jj = 0; jj < DVV; ++jj) {
        float zg = zrow[jj];
        float v  = orow[jj] * r * nw[jj] * (zg / (1.f + __expf(-zg)));
        onb[(size_t)idx * DVV + jj] = (__bf16)v;
    }
}

// ---------------- launcher ----------------
extern "C" void kernel_launch(void* const* d_in, const int* in_sizes, int n_in,
                              void* d_out, int out_size, void* d_ws, size_t ws_size,
                              hipStream_t stream) {
    const float* x       = (const float*)d_in[0];
    const float* W_qkv   = (const float*)d_in[1];
    const float* W_z     = (const float*)d_in[2];
    const float* W_a     = (const float*)d_in[3];
    const float* W_b     = (const float*)d_in[4];
    const float* W_out   = (const float*)d_in[5];
    const float* conv_w  = (const float*)d_in[6];
    const float* dt_bias = (const float*)d_in[7];
    const float* A_log   = (const float*)d_in[8];
    const float* norm_w  = (const float*)d_in[9];
    float* out = (float*)d_out;

    char* ws = (char*)d_ws;
    size_t off = 0;
    auto take = [&](size_t bytes) -> char* {
        char* p = ws + off;
        off = (off + bytes + 255) & ~(size_t)255;
        return p;
    };
    __bf16* xb     = (__bf16*)take((size_t)BTOT * DIMM * 2);
    __bf16* wqkvT  = (__bf16*)take((size_t)CONVD * DIMM * 2);
    __bf16* wzT    = (__bf16*)take((size_t)VALD * DIMM * 2);
    __bf16* woutT  = (__bf16*)take((size_t)DIMM * VALD * 2);
    __bf16* mixedb = (__bf16*)take((size_t)BTOT * CONVD * 2);
    float*  qc     = (float*) take((size_t)BTOT * CONVD * 4);
    float*  zbuf   = (float*) take((size_t)BTOT * VALD * 4);
    float*  gbuf   = (float*) take((size_t)BTOT * HVV * 4);
    float*  bbuf   = (float*) take((size_t)BTOT * HVV * 4);
    float*  obuf   = (float*) take((size_t)BTOT * VALD * 4);
    __bf16* onb    = (__bf16*)take((size_t)BTOT * VALD * 2);
    (void)ws_size; (void)in_sizes; (void)n_in; (void)out_size;

    const int TB = 256;
    auto blocks = [](size_t n, int tb) { return (unsigned)((n + tb - 1) / tb); };

    // 1) casts
    gdn_cast_bf16<<<blocks((size_t)BTOT * DIMM, TB), TB, 0, stream>>>(x, xb, (size_t)BTOT * DIMM);
    gdn_castT_bf16<<<blocks((size_t)DIMM * CONVD, TB), TB, 0, stream>>>(W_qkv, wqkvT, DIMM, CONVD);
    gdn_castT_bf16<<<blocks((size_t)DIMM * VALD, TB), TB, 0, stream>>>(W_z, wzT, DIMM, VALD);
    gdn_castT_bf16<<<blocks((size_t)VALD * DIMM, TB), TB, 0, stream>>>(W_out, woutT, VALD, DIMM);

    // 2) projections (WMMA GEMMs, 128x128 block tiles)
    gdn_gemm_bf16<__bf16><<<dim3(CONVD / 128, BTOT / 128), TB, 0, stream>>>(
        xb, wqkvT, mixedb, BTOT, CONVD, DIMM);
    gdn_gemm_bf16<float><<<dim3(VALD / 128, BTOT / 128), TB, 0, stream>>>(
        xb, wzT, zbuf, BTOT, VALD, DIMM);

    // 3) conv + silu, q/k l2norm, gates
    gdn_conv_silu<<<blocks((size_t)BTOT * CONVD, TB), TB, 0, stream>>>(mixedb, conv_w, qc);
    gdn_prep_qk<<<blocks((size_t)BTOT * HKK, TB), TB, 0, stream>>>(qc);
    gdn_gate<<<blocks((size_t)BTOT * HVV, TB), TB, 0, stream>>>(x, W_a, W_b, dt_bias, A_log,
                                                               gbuf, bbuf);

    // 4) sequential delta-rule scan (TDM double-buffered prefetch)
    gdn_scan<<<BB * HVV, DKK, 0, stream>>>(qc, gbuf, bbuf, obuf);

    // 5) gated RMSNorm * silu(z) -> bf16
    gdn_norm_gate<<<blocks((size_t)BTOT * HVV, TB), TB, 0, stream>>>(obuf, zbuf, norm_w, onb);

    // 6) output projection (WMMA GEMM) -> fp32 result
    gdn_gemm_bf16<float><<<dim3(DIMM / 128, BTOT / 128), TB, 0, stream>>>(
        onb, woutT, out, BTOT, DIMM, VALD);
}
